// Model_29798483099757
// MI455X (gfx1250) — matrix operand order repaired and checked by reading, not compiled, not run on hardware
//
#include <hip/hip_runtime.h>
#include <hip/hip_bf16.h>
#include <math.h>

// ---------------- model dims ----------------
#define B_   4
#define T_   6
#define N_   1024
#define CS_  48
#define GS_  32
#define H_   270
#define AT_  32
#define DA_  64
#define E_   32
#define L_   32
#define ED_  400
#define OUT_ 4000
#define INV_SQRT_AT 0.17677669529663687f

typedef __attribute__((ext_vector_type(16))) _Float16 v16h;
typedef __attribute__((ext_vector_type(8)))  float    v8f;

__device__ __forceinline__ float sigf(float x) { return 1.f / (1.f + __expf(-x)); }

// =====================================================================
// Generic WMMA GEMM:  C = act( alpha * A(f16) @ B(f16) + beta*C + bias )
// A: M x K row-major (LDA), B: K x Nc row-major (LDB), C row-major (ldc)
// M % (16*MB) == 0, Nc % 16 == 0, K % 32 == 0 (buffers pre-padded, zeros)
// LDA/LDB are template constants so every fragment gather folds into the
// 24-bit VMEM immediate-offset field: one A base pointer + one B base
// pointer are the only per-iteration induction variables (no spills).
// One wave32 owns an MB*16 x 16 C block: the B fragment is loaded once per
// K-step and reused across MB A fragments -> MB back-to-back v_wmma issues.
// act: 0=none 1=leaky(0.01) 2=tanh 3=sigmoid
// =====================================================================
template <int MB, int LDA, int LDB>
__global__ void wmma_gemm_kernel(const _Float16* __restrict__ A,
                                 const _Float16* __restrict__ Bm,
                                 void* __restrict__ Cv,
                                 const float* __restrict__ bias,
                                 float alpha, int beta,
                                 int M, int K, int Nc, int ldc,
                                 long long sA, long long sB, long long sC,
                                 int act, int out_f16)
{
    const int NT  = Nc >> 4;
    const int MTB = (M >> 4) / MB;               // M super-tiles
    int wave = blockIdx.x * (blockDim.x >> 5) + (threadIdx.x >> 5);
    if (wave >= MTB * NT) return;                // wave-uniform exit
    const int mtb = wave / NT;
    const int nt  = wave % NT;
    const int lane = threadIdx.x & 31;
    const int g    = lane >> 4;                  // lane half-group
    const int mr   = lane & 15;                  // row (A) / col (B,C)

    // A base: row (mtb*MB*16 + mr), plus the lane-group K offset g*16 folded in.
    const _Float16* Ap = A + (size_t)blockIdx.z * sA
                           + (size_t)(mtb * MB * 16 + mr) * LDA + g * 16;
    // B base: col (nt*16 + mr), plus the lane-group row offset g*16 folded in.
    const _Float16* Bp = Bm + (size_t)blockIdx.z * sB
                            + (size_t)(g * 16) * LDB + nt * 16 + mr;

    v8f acc[MB];
#pragma unroll
    for (int m = 0; m < MB; ++m) acc[m] = (v8f){};

#pragma unroll 2
    for (int k0 = 0; k0 < K; k0 += 32) {
        if (k0 + 64 < K) __builtin_prefetch(Ap + 64, 0, 1);
        // B fragment (shared across MB tiles): this code's contiguous K order, half h -> K = g*16 + h
        v16h b;
#pragma unroll
        for (int h = 0; h < 16; ++h)
            b[h] = Bp[h * LDB];                  // compile-time immediate offsets

#pragma unroll
        for (int m = 0; m < MB; ++m) {
            // A fragment: contiguous K order (same as B), half h -> K = g*16 + h
            // (g*16 already folded into Ap; offsets {0..7} and {8..15} are contiguous
            //  -> two global_load_b128 per fragment)
            v16h a;
#pragma unroll
            for (int h = 0; h < 8; ++h) {
                a[h]     = Ap[m * 16 * LDA + h];
                a[8 + h] = Ap[m * 16 * LDA + 8 + h];
            }
            acc[m] = __builtin_amdgcn_wmma_f32_16x16x32_f16(false, a, false, b,
                                                            (short)0, acc[m], false, false);
        }
        Ap += 32;
        Bp += (size_t)32 * LDB;
    }

    const int col = nt * 16 + mr;
#pragma unroll
    for (int m = 0; m < MB; ++m) {
#pragma unroll
        for (int r = 0; r < 8; ++r) {
            int row = (mtb * MB + m) * 16 + g * 8 + r;   // C layout: VGPR r -> M = r + g*8
            size_t ci = (size_t)blockIdx.z * sC + (size_t)row * ldc + col;
            float x = acc[m][r] * alpha;
            if (beta) x += ((const float*)Cv)[ci];
            if (bias) x += bias[col];
            if (act == 1)      x = x > 0.f ? x : 0.01f * x;
            else if (act == 2) x = tanhf(x);
            else if (act == 3) x = sigf(x);
            if (out_f16) ((_Float16*)Cv)[ci] = (_Float16)x;
            else         ((float*)Cv)[ci]    = x;
        }
    }
}

// -------------------- utility kernels --------------------
__global__ void cvt_pad_f16_kernel(const float* __restrict__ src, _Float16* __restrict__ dst,
                                   int rows, int cols, int prows, int pcols)
{
    int idx = blockIdx.x * blockDim.x + threadIdx.x;
    if (idx >= prows * pcols) return;
    int r = idx / pcols, c = idx % pcols;
    float v = (r < rows && c < cols) ? src[r * cols + c] : 0.f;
    dst[idx] = (_Float16)v;
}

__global__ void pad_bias_kernel(const float* __restrict__ src, float* __restrict__ dst, int n, int pn)
{
    int i = blockIdx.x * blockDim.x + threadIdx.x;
    if (i < pn) dst[i] = (i < n) ? src[i] : 0.f;
}

__global__ void zero_u32_kernel(unsigned* __restrict__ p, long long n)
{
    long long i = (long long)blockIdx.x * blockDim.x + threadIdx.x;
    if (i < n) p[i] = 0u;
}

// masked embeddings ce/ne -> f16 (B*2, N, 64) padded
__global__ void embed_kernel(const int* __restrict__ code_x, const int* __restrict__ neighbors,
                             const float* __restrict__ c_emb, const float* __restrict__ n_emb,
                             _Float16* __restrict__ cne_h, int t)
{
    int bn = blockIdx.x; int b = bn >> 10; int n = bn & 1023;
    int c = threadIdx.x;                 // 0..63
    float cm = (float)code_x[(b * T_ + t) * N_ + n];
    float nm = (float)neighbors[(b * T_ + t) * N_ + n];
    float ce = (c < CS_) ? cm * c_emb[n * CS_ + c] : 0.f;
    float ne = (c < CS_) ? nm * n_emb[n * CS_ + c] : 0.f;
    cne_h[((size_t)(b * 2 + 0) * N_ + n) * 64 + c] = (_Float16)ce;
    cne_h[((size_t)(b * 2 + 1) * N_ + n) * 64 + c] = (_Float16)ne;
}

// co_pre = ce + cm*(Ace+Ane), no_pre = ne + nm*(Ace+Ane) -> f16 padded
__global__ void combine_kernel(const int* __restrict__ code_x, const int* __restrict__ neighbors,
                               const float* __restrict__ c_emb, const float* __restrict__ n_emb,
                               const float* __restrict__ Acne, _Float16* __restrict__ pre_h, int t)
{
    int bn = blockIdx.x; int b = bn >> 10; int n = bn & 1023;
    int c = threadIdx.x;                 // 0..63
    float cm = (float)code_x[(b * T_ + t) * N_ + n];
    float nm = (float)neighbors[(b * T_ + t) * N_ + n];
    float cv = 0.f, nv = 0.f;
    if (c < CS_) {
        float Ace = Acne[((size_t)(b * 2 + 0) * N_ + n) * CS_ + c];
        float Ane = Acne[((size_t)(b * 2 + 1) * N_ + n) * CS_ + c];
        float s = Ace + Ane;
        cv = cm * c_emb[n * CS_ + c] + cm * s;
        nv = nm * n_emb[n * CS_ + c] + nm * s;
    }
    pre_h[((size_t)(b * 2 + 0) * N_ + n) * 64 + c] = (_Float16)cv;
    pre_h[((size_t)(b * 2 + 1) * N_ + n) * 64 + c] = (_Float16)nv;
}

// split co/no, build attention query input (m3 ? u_emb : no_prev)
__global__ void split_qin_kernel(const float* __restrict__ cono, const int* __restrict__ divided,
                                 const float* __restrict__ u_emb, const float* __restrict__ no_prev,
                                 _Float16* __restrict__ co_h, float* __restrict__ no_cur,
                                 _Float16* __restrict__ qin_h, int t)
{
    int bn = blockIdx.x; int b = bn >> 10; int n = bn & 1023;
    int c = threadIdx.x;                 // 0..31
    size_t row = (size_t)b * N_ + n;
    float co = cono[((size_t)(b * 2 + 0) * N_ + n) * GS_ + c];
    float no = cono[((size_t)(b * 2 + 1) * N_ + n) * GS_ + c];
    co_h[row * GS_ + c]  = (_Float16)co;
    no_cur[row * GS_ + c] = no;
    int m3 = divided[(((size_t)b * T_ + t) * N_ + n) * 3 + 2] > 0;
    float qi = m3 ? u_emb[n * GS_ + c] : no_prev[row * GS_ + c];
    qin_h[row * GS_ + c] = (_Float16)qi;
}

__global__ void ktrans_kernel(const float* __restrict__ kf, _Float16* __restrict__ kT_h)
{
    int bn = blockIdx.x; int b = bn >> 10; int n = bn & 1023;
    int a = threadIdx.x;                 // 0..31
    kT_h[((size_t)b * AT_ + a) * N_ + n] = (_Float16)kf[((size_t)b * N_ + n) * AT_ + a];
}

// masked softmax over keys j for one (b,i) row; writes f16 probabilities
__global__ void softmax_kernel(const float* __restrict__ gmat, const int* __restrict__ divided,
                               _Float16* __restrict__ score_h, int t)
{
    __shared__ float red[256];
    __shared__ int  redi[256];
    int b = blockIdx.x >> 10, i = blockIdx.x & 1023;
    int tid = threadIdx.x;
    const float* grow = gmat + ((size_t)b * N_ + i) * N_;
    _Float16* srow = score_h + ((size_t)b * N_ + i) * N_;

    float mx = -3.4e38f; int has = 0;
    for (int j = tid; j < N_; j += 256) {
        const int* d = divided + (((size_t)b * T_ + t) * N_ + j) * 3;
        int m23 = (d[1] > 0) | (d[2] > 0);
        has |= m23;
        if (m23) mx = fmaxf(mx, grow[j]);
    }
    red[tid] = mx; redi[tid] = has; __syncthreads();
    for (int s = 128; s > 0; s >>= 1) {
        if (tid < s) { red[tid] = fmaxf(red[tid], red[tid + s]); redi[tid] |= redi[tid + s]; }
        __syncthreads();
    }
    mx = red[0]; has = redi[0]; __syncthreads();

    if (has) {
        float sum = 0.f;
        for (int j = tid; j < N_; j += 256) {
            const int* d = divided + (((size_t)b * T_ + t) * N_ + j) * 3;
            if ((d[1] > 0) | (d[2] > 0)) sum += __expf(grow[j] - mx);
        }
        red[tid] = sum; __syncthreads();
        for (int s = 128; s > 0; s >>= 1) { if (tid < s) red[tid] += red[tid + s]; __syncthreads(); }
        float inv = 1.f / red[0];
        for (int j = tid; j < N_; j += 256) {
            const int* d = divided + (((size_t)b * T_ + t) * N_ + j) * 3;
            int m23 = (d[1] > 0) | (d[2] > 0);
            float p = m23 ? __expf(grow[j] - mx) * inv : 0.f;
            srow[j] = (_Float16)p;
        }
    } else {
        _Float16 u = (_Float16)(1.f / (float)N_);
        for (int j = tid; j < N_; j += 256) srow[j] = u;
    }
}

// GRU cell elementwise: hm1 = (1-z)*n + z*h_prev
__global__ void gru_kernel(const float* __restrict__ gi, const float* __restrict__ gh,
                           const float* __restrict__ bih, const float* __restrict__ bhh,
                           const float* __restrict__ hprev, float* __restrict__ hm1)
{
    long long idx = (long long)blockIdx.x * blockDim.x + threadIdx.x;
    if (idx >= (long long)(B_ * N_) * H_) return;
    size_t row = idx / H_; int j = idx % H_;
    const float* gir = gi + row * 816;
    const float* ghr = gh + row * 816;
    float r  = sigf(gir[j]       + bih[j]       + ghr[j]       + bhh[j]);
    float z  = sigf(gir[270 + j] + bih[270 + j] + ghr[270 + j] + bhh[270 + j]);
    float hn = ghr[540 + j] + bhh[540 + j];
    float n  = tanhf(gir[540 + j] + bih[540 + j] + r * hn);
    hm1[idx] = (1.f - z) * n + z * hprev[idx];
}

// carry select: h_new = m23&&t>0 ? h_m23 : (m1 ? h_m1 : 0); writes f32 + padded f16
__global__ void carry_kernel(const int* __restrict__ divided, const float* __restrict__ hm1,
                             const float* __restrict__ hm23, float* __restrict__ h,
                             _Float16* __restrict__ h_h, int t)
{
    int row = blockIdx.x; int b = row >> 10; int n = row & 1023;
    int j = threadIdx.x;                 // 0..287
    const int* d = divided + (((size_t)b * T_ + t) * N_ + n) * 3;
    int m1  = d[0] > 0;
    int m23 = (d[1] > 0) | (d[2] > 0);
    float v = 0.f;
    if (j < H_) {
        if (m1) v = hm1[(size_t)row * H_ + j];
        if (t > 0 && m23) v = hm23[(size_t)row * 272 + j];
        h[(size_t)row * H_ + j] = v;
    }
    h_h[(size_t)row * 288 + j] = (_Float16)v;
}

// masked max over nodes at last step: out_last = out_m1 + out_m23
__global__ void outlast_kernel(const int* __restrict__ divided, const float* __restrict__ hm1,
                               const float* __restrict__ hm23, float* __restrict__ out_last)
{
    int b = blockIdx.x; int j = threadIdx.x;
    if (j >= H_) return;
    float mx1 = -3.4e38f, mx23 = -3.4e38f; int has1 = 0, has23 = 0;
    for (int n = 0; n < N_; ++n) {
        const int* d = divided + (((size_t)b * T_ + (T_ - 1)) * N_ + n) * 3;
        if (d[0] > 0) { has1 = 1;  mx1  = fmaxf(mx1,  hm1[((size_t)b * N_ + n) * H_ + j]); }
        if ((d[1] > 0) | (d[2] > 0)) { has23 = 1; mx23 = fmaxf(mx23, hm23[((size_t)b * N_ + n) * 272 + j]); }
    }
    out_last[b * H_ + j] = (has1 ? mx1 : 0.f) + (has23 ? mx23 : 0.f);
}

// gather embedded event sequence -> f16 (L, B*E, 416) padded
__global__ void xgather_kernel(const int* __restrict__ events, const float* __restrict__ Eemb,
                               _Float16* __restrict__ x_all)
{
    int blk = blockIdx.x;                // l*128 + be
    int l = blk >> 7; int be = blk & 127; int b = be >> 5; int e = be & 31;
    int d = threadIdx.x;                 // 0..415
    int ev = events[(((size_t)b * T_ + (T_ - 1)) * E_ + e) * L_ + l];
    float v = (d < ED_) ? Eemb[(size_t)ev * ED_ + d] : 0.f;
    x_all[((size_t)l * 128 + be) * 416 + d] = (_Float16)v;
}

// LSTM cell elementwise
__global__ void lstm_kernel(const float* __restrict__ lg, const float* __restrict__ lb,
                            float* __restrict__ lc, float* __restrict__ lh,
                            _Float16* __restrict__ lh_h)
{
    int be = blockIdx.x; int j = threadIdx.x;   // 0..287
    if (j < H_) {
        const float* gr = lg + (size_t)be * 1088;
        float i  = sigf(gr[j]       + lb[j]);
        float f  = sigf(gr[270 + j] + lb[270 + j]);
        float gg = tanhf(gr[540 + j] + lb[540 + j]);
        float o  = sigf(gr[810 + j] + lb[810 + j]);
        float c  = f * lc[be * H_ + j] + i * gg;
        lc[be * H_ + j] = c;
        float hv = o * tanhf(c);
        lh[be * H_ + j] = hv;
        lh_h[(size_t)be * 288 + j] = (_Float16)hv;
    } else {
        lh_h[(size_t)be * 288 + j] = (_Float16)0.f;
    }
}

// dp-attention scores: vu[b,i] = ((x_i @ Wd) + bd) . ctx   (x_0=out_last, x_{1..32}=h_ev)
__global__ void vu_kernel(const float* __restrict__ out_last, const float* __restrict__ lh,
                          const float* __restrict__ Wd, const float* __restrict__ bd,
                          const float* __restrict__ ctx, float* __restrict__ vu)
{
    __shared__ float red[64];
    int bi = blockIdx.x; int b = bi / 33; int i = bi % 33;
    int da = threadIdx.x;                // 0..63
    const float* x = (i == 0) ? (out_last + b * H_) : (lh + ((size_t)b * E_ + (i - 1)) * H_);
    float acc = bd[da];
    for (int hh = 0; hh < H_; ++hh) acc += x[hh] * Wd[hh * DA_ + da];
    acc *= ctx[da];
    red[da] = acc; __syncthreads();
    for (int s = 32; s > 0; s >>= 1) { if (da < s) red[da] += red[da + s]; __syncthreads(); }
    if (da == 0) vu[bi] = red[0];
}

// o1 = sum_i softmax(vu)_i * x_i  -> f16 (16 x 288) padded (note o2 == o1)
__global__ void o1_kernel(const float* __restrict__ vu, const float* __restrict__ out_last,
                          const float* __restrict__ lh, _Float16* __restrict__ o1_h)
{
    int b = blockIdx.x;                  // 0..15 (rows >=4 are zero pad)
    int j = threadIdx.x;                 // 0..287
    float val = 0.f;
    if (b < B_ && j < H_) {
        const float* v = vu + b * 33;
        float mx = v[0];
        for (int i = 1; i < 33; ++i) mx = fmaxf(mx, v[i]);
        float sum = 0.f;
        for (int i = 0; i < 33; ++i) sum += __expf(v[i] - mx);
        float inv = 1.f / sum;
        for (int i = 0; i < 33; ++i) {
            float s = __expf(v[i] - mx) * inv;
            const float* x = (i == 0) ? (out_last + b * H_) : (lh + ((size_t)b * E_ + (i - 1)) * H_);
            val += s * x[j];
        }
    }
    o1_h[(size_t)b * 288 + j] = (_Float16)val;
}

__global__ void copy_out_kernel(const float* __restrict__ src, float* __restrict__ dst, int n)
{
    int i = blockIdx.x * blockDim.x + threadIdx.x;
    if (i < n) dst[i] = src[i];
}

// -------------------- host-side GEMM dispatch --------------------
static void gemm(hipStream_t s, const _Float16* A, const _Float16* Bm, void* C,
                 const float* bias, float alpha, int beta,
                 int M, int K, int Nc, int lda, int ldb, int ldc,
                 long long sA, long long sB, long long sC, int batch, int act, int of16)
{
    const int wpb = 4;                   // 4 waves = 128 threads
    int MT = M / 16;
    int mb = ((MT & 3) == 0) ? 4 : 1;
    int tiles = (MT / mb) * (Nc / 16);
    dim3 grid((tiles + wpb - 1) / wpb, 1, batch);

#define GEMM_CASE(MBv, LDAv, LDBv)                                                   \
    if (mb == MBv && lda == LDAv && ldb == LDBv) {                                   \
        wmma_gemm_kernel<MBv, LDAv, LDBv><<<grid, wpb * 32, 0, s>>>(                 \
            A, Bm, C, bias, alpha, beta, M, K, Nc, ldc, sA, sB, sC, act, of16);      \
        return;                                                                      \
    }
    GEMM_CASE(4, 1024,   64)   // adj @ {ce,ne}
    GEMM_CASE(4,   64,   32)   // pre @ Wg
    GEMM_CASE(4,   32,   32)   // qin@Wq, co@Wk
    GEMM_CASE(4,   32,  272)   // co @ Wv
    GEMM_CASE(4,   32, 1024)   // q @ kT (logits)
    GEMM_CASE(4, 1024,  272)   // score @ v
    GEMM_CASE(4,   32,  816)   // co @ gru_Wih
    GEMM_CASE(4,  288,  816)   // h @ gru_Whh
    GEMM_CASE(4,  416, 1088)   // x @ lstm_Wih
    GEMM_CASE(4,  288, 1088)   // h @ lstm_Whh
    GEMM_CASE(1,  288, 4000)   // o1 @ Wc
#undef GEMM_CASE
}

static void cvt_pad(hipStream_t s, const float* src, _Float16* dst, int r, int c, int pr, int pc)
{
    int n = pr * pc;
    cvt_pad_f16_kernel<<<(n + 255) / 256, 256, 0, s>>>(src, dst, r, c, pr, pc);
}

static void zero_ws(hipStream_t s, void* p, size_t bytes)
{
    long long n = (long long)(bytes / 4);
    zero_u32_kernel<<<(unsigned)((n + 255) / 256), 256, 0, s>>>((unsigned*)p, n);
}

extern "C" void kernel_launch(void* const* d_in, const int* in_sizes, int n_in,
                              void* d_out, int out_size, void* d_ws, size_t ws_size,
                              hipStream_t stream)
{
    (void)in_sizes; (void)n_in; (void)out_size; (void)ws_size;
    // ---- inputs (setup_inputs order) ----
    const int*   code_x    = (const int*)d_in[0];
    const int*   divided   = (const int*)d_in[1];
    const int*   neighbors = (const int*)d_in[2];
    const int*   events    = (const int*)d_in[4];
    const float* c_emb     = (const float*)d_in[5];
    const float* n_emb     = (const float*)d_in[6];
    const float* u_emb     = (const float*)d_in[7];
    const float* adj       = (const float*)d_in[8];
    const float* Wg        = (const float*)d_in[9];
    const float* bg        = (const float*)d_in[10];
    const float* gWih      = (const float*)d_in[11];
    const float* gWhh      = (const float*)d_in[12];
    const float* gbih      = (const float*)d_in[13];
    const float* gbhh      = (const float*)d_in[14];
    const float* Wq        = (const float*)d_in[15];
    const float* bq        = (const float*)d_in[16];
    const float* Wk        = (const float*)d_in[17];
    const float* bk        = (const float*)d_in[18];
    const float* Wv        = (const float*)d_in[19];
    const float* bv        = (const float*)d_in[20];
    const float* Wd        = (const float*)d_in[21];
    const float* bd        = (const float*)d_in[22];
    const float* ctx       = (const float*)d_in[23];
    const float* Eemb      = (const float*)d_in[24];
    const float* lWih      = (const float*)d_in[25];
    const float* lWhh      = (const float*)d_in[26];
    const float* lb        = (const float*)d_in[27];
    const float* Wc        = (const float*)d_in[28];
    const float* bc        = (const float*)d_in[29];
    float* out = (float*)d_out;

    // ---- workspace carve ----
    char* wp = (char*)d_ws;
    auto alloc = [&](size_t bytes) -> void* {
        void* r = (void*)wp;
        wp += (bytes + 255) & ~(size_t)255;
        return r;
    };
    _Float16* adj_h   = (_Float16*)alloc((size_t)N_ * N_ * 2);
    _Float16* Wg_h    = (_Float16*)alloc(64 * 32 * 2);
    _Float16* Wq_h    = (_Float16*)alloc(32 * 32 * 2);
    _Float16* Wk_h    = (_Float16*)alloc(32 * 32 * 2);
    _Float16* Wv_h    = (_Float16*)alloc(32 * 272 * 2);
    _Float16* gWih_h  = (_Float16*)alloc(32 * 816 * 2);
    _Float16* gWhh_h  = (_Float16*)alloc(288 * 816 * 2);
    _Float16* lWih_h  = (_Float16*)alloc(416 * 1088 * 2);
    _Float16* lWhh_h  = (_Float16*)alloc(288 * 1088 * 2);
    _Float16* Wc_h    = (_Float16*)alloc(288 * 4000 * 2);
    float*    bv_p    = (float*)alloc(272 * 4);
    _Float16* cne_h   = (_Float16*)alloc((size_t)8 * N_ * 64 * 2);
    float*    Acne    = (float*)alloc((size_t)8 * N_ * 48 * 4);
    _Float16* pre_h   = (_Float16*)alloc((size_t)8 * N_ * 64 * 2);
    float*    cono    = (float*)alloc((size_t)8 * N_ * 32 * 4);
    _Float16* co_h    = (_Float16*)alloc((size_t)4096 * 32 * 2);
    float*    no0     = (float*)alloc((size_t)4096 * 32 * 4);
    float*    no1     = (float*)alloc((size_t)4096 * 32 * 4);
    _Float16* qin_h   = (_Float16*)alloc((size_t)4096 * 32 * 2);
    _Float16* q_h     = (_Float16*)alloc((size_t)4096 * 32 * 2);
    float*    kf      = (float*)alloc((size_t)4096 * 32 * 4);
    _Float16* kT_h    = (_Float16*)alloc((size_t)B_ * 32 * N_ * 2);
    _Float16* v_h     = (_Float16*)alloc((size_t)4096 * 272 * 2);
    float*    gbuf    = (float*)alloc((size_t)B_ * N_ * N_ * 4);
    _Float16* score_h = (_Float16*)alloc((size_t)B_ * N_ * N_ * 2);
    float*    hm23    = (float*)alloc((size_t)4096 * 272 * 4);
    float*    gi      = (float*)alloc((size_t)4096 * 816 * 4);
    float*    gh      = (float*)alloc((size_t)4096 * 816 * 4);
    float*    hm1     = (float*)alloc((size_t)4096 * 270 * 4);
    float*    hbuf    = (float*)alloc((size_t)4096 * 270 * 4);
    _Float16* h_h     = (_Float16*)alloc((size_t)4096 * 288 * 2);
    float*    out_last= (float*)alloc((size_t)B_ * H_ * 4);
    _Float16* x_all   = (_Float16*)alloc((size_t)32 * 128 * 416 * 2);
    float*    lg      = (float*)alloc((size_t)128 * 1088 * 4);
    float*    lcbuf   = (float*)alloc((size_t)128 * 270 * 4);
    float*    lhbuf   = (float*)alloc((size_t)128 * 270 * 4);
    _Float16* lh_h    = (_Float16*)alloc((size_t)128 * 288 * 2);
    float*    vu      = (float*)alloc((size_t)B_ * 33 * 4);
    _Float16* o1_h    = (_Float16*)alloc((size_t)16 * 288 * 2);
    float*    Cfin    = (float*)alloc((size_t)16 * 4000 * 4);

    // ---- once-per-launch precompute: f16 padded weights, zeroed state ----
    cvt_pad(stream, adj,  adj_h,  N_, N_, N_, N_);
    cvt_pad(stream, Wg,   Wg_h,   48, 32, 64, 32);
    cvt_pad(stream, Wq,   Wq_h,   32, 32, 32, 32);
    cvt_pad(stream, Wk,   Wk_h,   32, 32, 32, 32);
    cvt_pad(stream, Wv,   Wv_h,   32, 270, 32, 272);
    cvt_pad(stream, gWih, gWih_h, 32, 810, 32, 816);
    cvt_pad(stream, gWhh, gWhh_h, 270, 810, 288, 816);
    cvt_pad(stream, lWih, lWih_h, 400, 1080, 416, 1088);
    cvt_pad(stream, lWhh, lWhh_h, 270, 1080, 288, 1088);
    cvt_pad(stream, Wc,   Wc_h,   270, 4000, 288, 4000);
    pad_bias_kernel<<<2, 256, 0, stream>>>(bv, bv_p, 270, 272);
    zero_ws(stream, hbuf, (size_t)4096 * 270 * 4);
    zero_ws(stream, h_h,  (size_t)4096 * 288 * 2);
    zero_ws(stream, no0,  (size_t)4096 * 32 * 4);
    zero_ws(stream, no1,  (size_t)4096 * 32 * 4);
    zero_ws(stream, lcbuf,(size_t)128 * 270 * 4);
    zero_ws(stream, lh_h, (size_t)128 * 288 * 2);

    float* nobuf[2] = { no0, no1 };

    // ==================== visit scan: t = 0..T-1 ====================
    for (int t = 0; t < T_; ++t) {
        float* no_prev = nobuf[t & 1];
        float* no_cur  = nobuf[(t + 1) & 1];

        // graph conv: ce/ne, Ace/Ane = adj @ {ce,ne}, then Wg + leaky_relu
        embed_kernel<<<B_ * N_, 64, 0, stream>>>(code_x, neighbors, c_emb, n_emb, cne_h, t);
        gemm(stream, adj_h, cne_h, Acne, nullptr, 1.f, 0,
             N_, N_, 48, N_, 64, 48, 0, (long long)N_ * 64, (long long)N_ * 48, 8, 0, 0);
        combine_kernel<<<B_ * N_, 64, 0, stream>>>(code_x, neighbors, c_emb, n_emb, Acne, pre_h, t);
        gemm(stream, pre_h, Wg_h, cono, bg, 1.f, 0,
             8 * N_, 64, 32, 64, 32, 32, 0, 0, 0, 1, /*leaky*/1, 0);
        split_qin_kernel<<<B_ * N_, 32, 0, stream>>>(cono, divided, u_emb, no_prev,
                                                     co_h, no_cur, qin_h, t);

        if (t > 0) {   // attention branch is discarded at t==0 by the reference gating
            gemm(stream, qin_h, Wq_h, q_h, bq, 1.f, 0,
                 B_ * N_, 32, 32, 32, 32, 32, 0, 0, 0, 1, 0, /*f16*/1);
            gemm(stream, co_h, Wk_h, kf, bk, 1.f, 0,
                 B_ * N_, 32, 32, 32, 32, 32, 0, 0, 0, 1, 0, 0);
            ktrans_kernel<<<B_ * N_, 32, 0, stream>>>(kf, kT_h);
            gemm(stream, co_h, Wv_h, v_h, bv_p, 1.f, 0,
                 B_ * N_, 32, 272, 32, 272, 272, 0, 0, 0, 1, 0, /*f16*/1);
            // logits g = q @ k^T * inv_sqrt(AT)   (batched over B)
            gemm(stream, q_h, kT_h, gbuf, nullptr, INV_SQRT_AT, 0,
                 N_, 32, N_, 32, N_, N_,
                 (long long)N_ * 32, (long long)32 * N_, (long long)N_ * N_, B_, 0, 0);
            softmax_kernel<<<B_ * N_, 256, 0, stream>>>(gbuf, divided, score_h, t);
            // h_m23 = tanh(score @ v)
            gemm(stream, score_h, v_h, hm23, nullptr, 1.f, 0,
                 N_, N_, 272, N_, 272, 272,
                 (long long)N_ * N_, (long long)N_ * 272, (long long)N_ * 272, B_, /*tanh*/2, 0);
        }

        // GRU: gi = co@Wih, gh = h@Whh, then cell math
        gemm(stream, co_h, gWih_h, gi, nullptr, 1.f, 0,
             B_ * N_, 32, 816, 32, 816, 816, 0, 0, 0, 1, 0, 0);
        gemm(stream, h_h, gWhh_h, gh, nullptr, 1.f, 0,
             B_ * N_, 288, 816, 288, 816, 816, 0, 0, 0, 1, 0, 0);
        {
            long long n = (long long)(B_ * N_) * H_;
            gru_kernel<<<(unsigned)((n + 255) / 256), 256, 0, stream>>>(gi, gh, gbih, gbhh, hbuf, hm1);
        }
        carry_kernel<<<B_ * N_, 288, 0, stream>>>(divided, hm1, hm23, hbuf, h_h, t);
        if (t == T_ - 1)
            outlast_kernel<<<B_, 288, 0, stream>>>(divided, hm1, hm23, out_last);
    }

    // ==================== event LSTM: l = 0..L-1 ====================
    xgather_kernel<<<L_ * 128, 416, 0, stream>>>(events, Eemb, x_all);
    for (int l = 0; l < L_; ++l) {
        gemm(stream, x_all + (size_t)l * 128 * 416, lWih_h, lg, nullptr, 1.f, 0,
             128, 416, 1088, 416, 1088, 1088, 0, 0, 0, 1, 0, 0);
        gemm(stream, lh_h, lWhh_h, lg, nullptr, 1.f, /*accumulate*/1,
             128, 288, 1088, 288, 1088, 1088, 0, 0, 0, 1, 0, 0);
        lstm_kernel<<<128, 288, 0, stream>>>(lg, lb, lcbuf, lhbuf, lh_h);
    }

    // ==================== dp-attention head + classifier ====================
    vu_kernel<<<B_ * 33, 64, 0, stream>>>(out_last, lhbuf, Wd, bd, ctx, vu);
    o1_kernel<<<16, 288, 0, stream>>>(vu, out_last, lhbuf, o1_h);
    gemm(stream, o1_h, Wc_h, Cfin, bc, 1.f, 0,
         16, 288, 4000, 288, 4000, 4000, 0, 0, 0, 1, /*sigmoid*/3, 0);
    copy_out_kernel<<<(B_ * OUT_ + 255) / 256, 256, 0, stream>>>(Cfin, out, B_ * OUT_);
}
